// BidirectionalGRUEncoder_58832462021288
// MI455X (gfx1250) — compile-verified, matrix operand order checked by reading, not run on hardware
//
#include <hip/hip_runtime.h>
#include <hip/hip_bf16.h>

// ---------------- problem constants ----------------
constexpr int H    = 600;          // hidden size
constexpr int B    = 16;
constexpr int S    = 1024;
constexpr int T    = B * S;        // 16384 serial steps per direction
constexpr int C    = 509;          // context feature dim
constexpr int IN   = 512;          // GRU input size (3 + 509) == GEMM K
constexpr int G3H  = 1800;         // 3*H
constexpr int NWG  = 30;           // workgroups per direction (H/20)
constexpr int JC   = H / NWG;      // 20 hidden units owned per WG
constexpr int ROWS = 3 * JC;       // 60 W_hh rows resident in LDS per WG

// ---------------- workspace layout (bytes) ----------------
constexpr size_t OFF_GX  = 0;                                   // T*G3H f32   (~118 MB)
constexpr size_t OFF_XB  = OFF_GX + (size_t)T * G3H * 4;        // T*IN bf16   (~16.8 MB)
constexpr size_t OFF_WB  = OFF_XB + (size_t)T * IN * 2;         // G3H*IN bf16 (~1.8 MB)
constexpr size_t OFF_HB  = OFF_WB + (size_t)G3H * IN * 2;       // 2 dir * 2 buf * H f32
constexpr size_t OFF_CNT = OFF_HB + 2 * 2 * H * 4;              // 2 x u32 sync counters

typedef __attribute__((ext_vector_type(16))) __bf16 v16bf;
typedef __attribute__((ext_vector_type(8)))  float  v8f;

// ---------------- kernel 0: zero hidden-state buffers + sync counters ----------------
__global__ void k_init(float* hb, unsigned* cnt) {
  int i = blockIdx.x * blockDim.x + threadIdx.x;
  if (i < 2 * 2 * H) hb[i] = 0.f;
  if (i < 2) cnt[i] = 0u;
}

// ---------------- kernel 1: x = [tag_emb[tags] | context] -> bf16 ----------------
__global__ void k_build_x(const float* __restrict__ ctx, const int* __restrict__ tags,
                          const float* __restrict__ temb, __bf16* __restrict__ xb) {
  int i = blockIdx.x * 256 + threadIdx.x;        // over T*IN
  int m = i >> 9;                                 // IN == 512
  int c = i & 511;
  float v = (c < 3) ? temb[tags[m] * 3 + c] : ctx[m * C + (c - 3)];
  xb[i] = (__bf16)v;
}

// ---------------- kernel 2: W_ih -> bf16 ----------------
__global__ void k_conv_w(const float* __restrict__ w, __bf16* __restrict__ wb) {
  int i = blockIdx.x * 256 + threadIdx.x;
  if (i < G3H * IN) wb[i] = (__bf16)w[i];
}

// ---------------- kernel 3: WMMA GEMM  gx[T,1800] = x @ W_ih^T + b_ih ----------------
// One wave computes one 16x16 tile. A: 16x32 bf16 tile of x; B: 32x16 bf16 tile of
// W_ih^T (i.e. contiguous K-runs of W_ih rows). K = 512 -> 16 wmma per tile.
__global__ void __launch_bounds__(256) k_gemm_wmma(const __bf16* __restrict__ xb,
                                                   const __bf16* __restrict__ wb,
                                                   const float* __restrict__ bih,
                                                   float* __restrict__ gx) {
  constexpr int NT = (G3H + 15) / 16;            // 113 N-tiles (last partially masked)
  int wave = blockIdx.x * 8 + (threadIdx.x >> 5);
  int lane = threadIdx.x & 31;
  int mt = wave / NT, nt = wave % NT;
  int m0 = mt * 16, n0 = nt * 16;
  int g  = lane >> 4;                            // K-half select (wave32 A/B layout)
  int li = lane & 15;

  const __bf16* arow = xb + (size_t)(m0 + li) * IN;
  int nb = n0 + li; if (nb > G3H - 1) nb = G3H - 1;   // clamp OOB column loads
  const __bf16* brow = wb + (size_t)nb * IN;

  v8f acc = {};
#pragma unroll
  for (int kt = 0; kt < IN / 32; ++kt) {
    int k0 = kt * 32 + 8 * g;
    int k1 = kt * 32 + 16 + 8 * g;
    union { v16bf v; uint4 q[2]; } a, b;
    a.q[0] = *(const uint4*)(arow + k0);
    a.q[1] = *(const uint4*)(arow + k1);
    b.q[0] = *(const uint4*)(brow + k0);
    b.q[1] = *(const uint4*)(brow + k1);
    acc = __builtin_amdgcn_wmma_f32_16x16x32_bf16(
        /*neg_a=*/false, a.v, /*neg_b=*/false, b.v,
        /*c_mod=*/(short)0, acc, /*reuse_a=*/false, /*reuse_b=*/false);
  }

  int n = n0 + li;
  if (n < G3H) {
    float bias = bih[n];
#pragma unroll
    for (int r = 0; r < 8; ++r)
      gx[(size_t)(m0 + r + 8 * g) * G3H + n] = acc[r] + bias;   // D layout: M=r+8*g, N=lane%16
  }
}

// ---------------- kernel 4: persistent sharded GRU recurrence ----------------
// 2 directions x 30 WGs. Each WG keeps its 60 W_hh rows (r/z/n slices for its 20
// hidden units) in LDS for the whole 16384-step chain. Per step: 60 length-600
// dot products (4 lanes each), fused gate math, publish 20 floats, one
// device-scope release/acquire barrier across the 30 WGs of the direction.
__global__ void __launch_bounds__(256) k_gru(const float* __restrict__ gx,
                                             const float* __restrict__ whh,
                                             const float* __restrict__ bhh,
                                             float* hb, unsigned* cnt,
                                             float* __restrict__ out) {
  __shared__ float sW[ROWS * H];   // 144 KB of 320 KB WGP LDS
  __shared__ float sB[ROWS];
  __shared__ float hL[H];
  __shared__ float sd[64];

  const int tid = threadIdx.x;
  const int dir = blockIdx.x / NWG;
  const int wg  = blockIdx.x % NWG;
  const int j0  = wg * JC;
  float* hbd    = hb + dir * (2 * H);
  unsigned* cn  = cnt + dir;

  auto growf = [&](int rl) {                       // local row -> W_hh row
    int gate = rl / JC;                            // 0=r, 1=z, 2=n
    return gate * H + j0 + (rl - gate * JC);
  };
  for (int idx = tid; idx < ROWS * H; idx += 256) {
    int rl = idx / H, k = idx - rl * H;
    sW[idx] = whh[growf(rl) * H + k];
  }
  if (tid < ROWS) sB[tid] = bhh[growf(tid)];
  __syncthreads();

  const int row = tid >> 2, sub = tid & 3;         // 4 lanes per dot product
  for (int t = 0; t < T; ++t) {
    // fetch current hidden state with agent scope (bypass stale WGP cache)
    const float* hsrc = hbd + (t & 1) * H;
    for (int k = tid; k < H; k += 256)
      hL[k] = __hip_atomic_load(&hsrc[k], __ATOMIC_RELAXED, __HIP_MEMORY_SCOPE_AGENT);
    __syncthreads();

    if (row < ROWS) {
      const float* w  = sW + row * H + sub * 150;
      const float* hh = hL + sub * 150;
      float a = 0.f;
#pragma unroll 10
      for (int k = 0; k < 150; ++k) a = fmaf(w[k], hh[k], a);
      a += __shfl_xor(a, 1, 4);                    // reduce the 4-lane group
      a += __shfl_xor(a, 2, 4);
      if (sub == 0) sd[row] = a + sB[row];
    }
    __syncthreads();

    if (tid < JC) {
      int j = j0 + tid;
      int rrow = (dir == 0) ? t : ((t & ~(S - 1)) + (S - 1) - (t & (S - 1)));
      const float* gr = gx + (size_t)rrow * G3H;
      float r  = 1.f / (1.f + __expf(-(gr[j]         + sd[tid])));
      float z  = 1.f / (1.f + __expf(-(gr[H + j]     + sd[JC + tid])));
      float nn = tanhf(gr[2 * H + j] + r * sd[2 * JC + tid]);
      float hn = (1.f - z) * nn + z * hL[j];
      __hip_atomic_store(&hbd[((t + 1) & 1) * H + j], hn,
                         __ATOMIC_RELAXED, __HIP_MEMORY_SCOPE_AGENT);
      if (t == T - 1) out[dir * H + j] = hn;
      if (t + 1 < T) {                             // global_prefetch_b8 next gx row
        int nr = (dir == 0) ? t + 1
                            : (((t + 1) & ~(S - 1)) + (S - 1) - ((t + 1) & (S - 1)));
        __builtin_prefetch(gx + (size_t)nr * G3H + j, 0, 1);
      }
    }
    __syncthreads();
    if (tid == 0) {                                // cross-WG step barrier (per direction)
      __hip_atomic_fetch_add(cn, 1u, __ATOMIC_RELEASE, __HIP_MEMORY_SCOPE_AGENT);
      unsigned tgt = (unsigned)NWG * (unsigned)(t + 1);
      while (__hip_atomic_load(cn, __ATOMIC_ACQUIRE, __HIP_MEMORY_SCOPE_AGENT) < tgt)
        __builtin_amdgcn_s_sleep(1);
    }
    __syncthreads();
  }
}

// ---------------- host launcher ----------------
extern "C" void kernel_launch(void* const* d_in, const int* in_sizes, int n_in,
                              void* d_out, int out_size, void* d_ws, size_t ws_size,
                              hipStream_t stream) {
  const float* ctx  = (const float*)d_in[0];   // [16,1024,509]
  const int*   tags = (const int*)  d_in[1];   // [16,1024]
  const float* temb = (const float*)d_in[2];   // [3,3]
  const float* wih  = (const float*)d_in[3];   // [1800,512]
  const float* whh  = (const float*)d_in[4];   // [1800,600]
  const float* bih  = (const float*)d_in[5];   // [1800]
  const float* bhh  = (const float*)d_in[6];   // [1800]
  float* out = (float*)d_out;                  // [1200]

  char* ws = (char*)d_ws;
  float*    gx  = (float*)   (ws + OFF_GX);
  __bf16*   xb  = (__bf16*)  (ws + OFF_XB);
  __bf16*   wb  = (__bf16*)  (ws + OFF_WB);
  float*    hbuf= (float*)   (ws + OFF_HB);
  unsigned* cnt = (unsigned*)(ws + OFF_CNT);

  k_init   <<<10, 256, 0, stream>>>(hbuf, cnt);
  k_build_x<<<(T * IN) / 256, 256, 0, stream>>>(ctx, tags, temb, xb);
  k_conv_w <<<(G3H * IN + 255) / 256, 256, 0, stream>>>(wih, wb);

  constexpr int NT = (G3H + 15) / 16;          // 113
  int waves = (T / 16) * NT;                   // 115712 tiles, 8 waves/block
  k_gemm_wmma<<<waves / 8, 256, 0, stream>>>(xb, wb, bih, gx);

  k_gru<<<2 * NWG, 256, 0, stream>>>(gx, whh, bhh, hbuf, cnt, out);
}
